// CFConv_65678639891014
// MI455X (gfx1250) — matrix operand order repaired and checked by reading, not compiled, not run on hardware
//
#include <hip/hip_runtime.h>
#include <hip/hip_bf16.h>
#include <stdint.h>

#define D_IN   64
#define D_EDGE 32
#define D_HID  128

typedef __attribute__((ext_vector_type(16))) _Float16 v16h;
typedef __attribute__((ext_vector_type(8)))  float    v8f;

union Frag16 { v16h v; uint32_t u[8]; };

__device__ __forceinline__ uint32_t pack_h2(float a, float b) {
    union { _Float16 h[2]; uint32_t u; } p;
    p.h[0] = (_Float16)a; p.h[1] = (_Float16)b;
    return p.u;
}

// branch-free tanh: 1 - 2/(exp(2x)+1); v_exp_f32 is TRANS (co-executes w/ VALU)
__device__ __forceinline__ float fast_tanh(float x) {
    float e = __builtin_amdgcn_exp2f(x * 2.88539008177793f);   // exp(2x)
    return 1.0f - 2.0f * __builtin_amdgcn_rcpf(e + 1.0f);
}

// monotone float<->uint encoding so unsigned atomicMax == float max
__device__ __forceinline__ uint32_t fenc(float f) {
    uint32_t u = __float_as_uint(f);
    return (u & 0x80000000u) ? ~u : (u | 0x80000000u);
}
__device__ __forceinline__ float fdec(uint32_t k) {
    uint32_t u = (k & 0x80000000u) ? (k & 0x7FFFFFFFu) : ~k;
    return __uint_as_float(u);
}

// ---------------------------------------------------------------------------
// Repack W1 [32,128] and W2 [128,64] (f32 row-major) into f16 WMMA B-fragment
// order, in ONE contiguous pool: [0,2048) = W1 (8 N-tiles), [2048,6144) = W2
// (4 N-tiles x 4 K-chunks). B layout (16-bit, K=32 chunk, wave32): VGPR v,
// lane l: N = 16*tile + (l&15), K = (v<4 ? 0 : 16) + (l>=16 ? 8 : 0) + 2*(v&3) (packed pair K,K+1)
// ---------------------------------------------------------------------------
__global__ void prep_frags(const float* __restrict__ W1, const float* __restrict__ W2,
                           uint32_t* __restrict__ wfrag) {
    int idx = blockIdx.x * blockDim.x + threadIdx.x;
    if (idx < 2048) {
        int t = idx >> 8, v = (idx >> 5) & 7, l = idx & 31;
        int n = t * 16 + (l & 15);
        int k = ((v & 4) << 2) + ((l & 16) ? 8 : 0) + ((v & 3) << 1);
        wfrag[idx] = pack_h2(W1[k * D_HID + n], W1[(k + 1) * D_HID + n]);
    }
    if (idx < 4096) {
        int f  = idx >> 8;            // f = t2*4 + kc
        int t2 = f >> 2, kc = f & 3;
        int v = (idx >> 5) & 7, l = idx & 31;
        int n = t2 * 16 + (l & 15);
        int k = kc * 32 + ((v & 4) << 2) + ((l & 16) ? 8 : 0) + ((v & 3) << 1);
        wfrag[2048 + idx] = pack_h2(W2[k * D_IN + n], W2[(k + 1) * D_IN + n]);
    }
}

__global__ void init_buffers(float* __restrict__ out, uint32_t* __restrict__ segmax,
                             float* __restrict__ denom, int n_nodes) {
    int idx = blockIdx.x * blockDim.x + threadIdx.x;
    if (idx < n_nodes * D_IN) out[idx] = 0.0f;
    if (idx < n_nodes) { segmax[idx] = 0u; denom[idx] = 0.0f; }
}

// ---------------------------------------------------------------------------
// One wave per 16-edge tile. Weight fragments staged once per block in LDS.
//   GEMM1 (16x128 = 8 WMMAs, K=32) -> fast_tanh -> f16 tile in LDS
//   GEMM2 (16x64 over K=128 = 16 WMMAs), A-fragments re-read from LDS
//   epilogue: gather x[source], stage f16 messages in LDS, coalesced b128
//             store-out, attention scores via 16-lane shfl reduction
// A layout (16-bit 16x32): lane l -> M=(l&15); VGPR v -> K pair at
//   (v<4?0:16) + (l>=16?8:0) + 2*(v&3)
// C/D layout: lane l -> N=(l&15); VGPR r -> M = r + (l>=16?8:0)
// ---------------------------------------------------------------------------
__global__ void __launch_bounds__(256, 2)
edge_mlp(const float* __restrict__ x,
         const int* __restrict__ src,
         const float* __restrict__ edge_attr,
         const uint32_t* __restrict__ wfrag,
         const float* __restrict__ b1,
         const float* __restrict__ b2,
         const float* __restrict__ attn_v,
         _Float16* __restrict__ msgs,
         float* __restrict__ scores,
         int n_edges) {
    __shared__ __align__(16) _Float16 hbuf[8 * 16 * D_HID];  // 32 KB, 4 KB/wave
    __shared__ __align__(16) uint32_t swbuf[6144];           // 24 KB weight frags

    // ---- stage W1/W2 fragments once per block (coalesced b128) ----
    {
        const uint4* g4 = (const uint4*)wfrag;
        uint4* s4 = (uint4*)swbuf;
        #pragma unroll
        for (int i = 0; i < 6; ++i) s4[i * 256 + threadIdx.x] = g4[i * 256 + threadIdx.x];
    }
    __syncthreads();

    const int wave = threadIdx.x >> 5;
    const int lane = threadIdx.x & 31;
    const int tile = blockIdx.x * 8 + wave;
    const int eb   = tile * 16;
    const bool active = (eb < n_edges);
    const int row = lane & 15;        // A: M row ; C/D: N column
    const int hi  = (lane >> 4) & 1;  // lane half

    _Float16* hmine = hbuf + wave * (16 * D_HID);

    if (active) {
        // ---- GEMM1: h = tanh(edge_attr @ W1 + b1) ----
        Frag16 A;
        int ea_row = eb + row;
        if (ea_row >= n_edges) ea_row = n_edges - 1;
        const float* ea = edge_attr + (long)ea_row * D_EDGE;
        if (eb + 16 + row < n_edges)
            __builtin_prefetch(edge_attr + (long)(eb + 16 + row) * D_EDGE, 0, 1);
        #pragma unroll
        for (int v = 0; v < 8; ++v) {
            int k = ((v & 4) << 2) + hi * 8 + ((v & 3) << 1);
            A.u[v] = pack_h2(ea[k], ea[k + 1]);
        }
        #pragma unroll
        for (int t = 0; t < 8; ++t) {
            Frag16 B;
            const uint32_t* bp = swbuf + t * 256 + lane;
            #pragma unroll
            for (int v = 0; v < 8; ++v) B.u[v] = bp[v * 32];
            v8f acc = {};
            acc = __builtin_amdgcn_wmma_f32_16x16x32_f16(
                false, A.v, false, B.v, (short)0, acc, false, false);
            float b1v = b1[t * 16 + row];
            #pragma unroll
            for (int r = 0; r < 8; ++r) {
                float h = fast_tanh(acc[r] + b1v);
                hmine[(r + hi * 8) * D_HID + t * 16 + row] = (_Float16)h;
            }
        }
    }
    __syncthreads();

    if (active) {
        // ---- GEMM2: edge_weights = h @ W2 + b2 ----
        v8f acc2[4];
        #pragma unroll
        for (int t2 = 0; t2 < 4; ++t2) acc2[t2] = (v8f){0,0,0,0,0,0,0,0};

        #pragma unroll
        for (int kc = 0; kc < 4; ++kc) {
            Frag16 A2;
            const _Float16* hr = hmine + row * D_HID + kc * 32;
            #pragma unroll
            for (int v = 0; v < 8; ++v) {
                int k = ((v & 4) << 2) + hi * 8 + ((v & 3) << 1);
                A2.u[v] = *(const uint32_t*)(hr + k);   // 4B-aligned LDS pair
            }
            #pragma unroll
            for (int t2 = 0; t2 < 4; ++t2) {
                Frag16 B2;
                const uint32_t* bp = swbuf + 2048 + (t2 * 4 + kc) * 256 + lane;
                #pragma unroll
                for (int v = 0; v < 8; ++v) B2.u[v] = bp[v * 32];
                acc2[t2] = __builtin_amdgcn_wmma_f32_16x16x32_f16(
                    false, A2.v, false, B2.v, (short)0, acc2[t2], false, false);
            }
        }

        // ---- epilogue: messages = x[source] * ew ; scores = messages @ attn_v
        int sidx = eb + row;
        if (sidx >= n_edges) sidx = n_edges - 1;
        int sfull = src[sidx];                       // one gather, bcast via shfl

        _Float16* mbuf = hmine;                      // reuse wave's LDS tile (2KB)
        float p[8];
        #pragma unroll
        for (int r = 0; r < 8; ++r) p[r] = 0.0f;
        #pragma unroll
        for (int t2 = 0; t2 < 4; ++t2) {
            int n = t2 * 16 + row;
            float b2v = b2[n];
            float av  = attn_v[n];
            #pragma unroll
            for (int r = 0; r < 8; ++r) {
                int s = __shfl(sfull, r + 8 * hi, 16);
                float ew  = acc2[t2][r] + b2v;
                float msg = x[(long)s * D_IN + n] * ew;
                mbuf[(r + 8 * hi) * D_IN + n] = (_Float16)msg;
                p[r] += msg * av;
            }
        }
        __builtin_amdgcn_wave_barrier();             // keep LDS stores before copy

        // coalesced f16 message tile store: 16 edges x 64 ch = 2KB contiguous
        if (eb + 16 <= n_edges) {
            const uint4* mb4 = (const uint4*)mbuf;
            uint4* mg = (uint4*)(msgs + (size_t)eb * D_IN);
            #pragma unroll
            for (int i = 0; i < 4; ++i) mg[i * 32 + lane] = mb4[i * 32 + lane];
        } else {
            for (int m = 0; m < 16; ++m) {
                int e2 = eb + m;
                if (e2 < n_edges)
                    ((uint32_t*)(msgs + (size_t)e2 * D_IN))[lane] =
                        ((const uint32_t*)(mbuf + m * D_IN))[lane];
            }
        }

        // reduce each row's score partial over its 16-lane segment (wave32)
        #pragma unroll
        for (int r = 0; r < 8; ++r) {
            float vsum = p[r];
            #pragma unroll
            for (int mask = 8; mask >= 1; mask >>= 1)
                vsum += __shfl_xor(vsum, mask, 16);
            if (row == r) {
                int e2 = eb + hi * 8 + r;
                if (e2 < n_edges) scores[e2] = vsum;
            }
        }
    }
}

__global__ void seg_max_k(const float* __restrict__ scores, const int* __restrict__ tgt,
                          uint32_t* __restrict__ segmax, int n_edges) {
    int e = blockIdx.x * blockDim.x + threadIdx.x;
    if (e < n_edges) atomicMax(&segmax[tgt[e]], fenc(scores[e]));
}

__global__ void denom_k(const float* __restrict__ scores, const int* __restrict__ tgt,
                        const uint32_t* __restrict__ segmax,
                        float* __restrict__ exbuf, float* __restrict__ denom, int n_edges) {
    int e = blockIdx.x * blockDim.x + threadIdx.x;
    if (e < n_edges) {
        int t = tgt[e];
        float ex = __expf(scores[e] - fdec(segmax[t]));
        exbuf[e] = ex;
        atomicAdd(&denom[t], ex);
    }
}

__global__ void scatter_k(const _Float16* __restrict__ msgs, const int* __restrict__ tgt,
                          const float* __restrict__ exbuf, const float* __restrict__ denom,
                          float* __restrict__ out, float* __restrict__ attn_out, long total) {
    long idx = (long)blockIdx.x * blockDim.x + threadIdx.x;
    if (idx < total) {
        int e = (int)(idx >> 6);
        int c = (int)(idx & 63);
        int t = tgt[e];
        float aw  = exbuf[e] / denom[t];
        float msg = (float)msgs[idx];
        atomicAdd(&out[(long)t * D_IN + c], msg * aw);
        if (c == 0) attn_out[e] = aw;
    }
}

extern "C" void kernel_launch(void* const* d_in, const int* in_sizes, int n_in,
                              void* d_out, int out_size, void* d_ws, size_t ws_size,
                              hipStream_t stream) {
    const float* x          = (const float*)d_in[0];   // [N,64]
    const int*   edge_index = (const int*)  d_in[1];   // [2,E]
    const float* edge_attr  = (const float*)d_in[2];   // [E,32]
    const float* W1 = (const float*)d_in[3];           // [32,128]
    const float* b1 = (const float*)d_in[4];           // [128]
    const float* W2 = (const float*)d_in[5];           // [128,64]
    const float* b2 = (const float*)d_in[6];           // [64]
    const float* attn_v = (const float*)d_in[7];       // [64,1]

    const int n_nodes = in_sizes[0] / D_IN;
    const int n_edges = in_sizes[2] / D_EDGE;
    const int* src = edge_index;
    const int* tgt = edge_index + n_edges;

    // workspace carve-out (~110 MB total; dominant: f16 messages)
    char* ws = (char*)d_ws;
    size_t off = 0;
    auto carve = [&](size_t bytes) -> void* {
        off = (off + 255) & ~(size_t)255;
        void* p = ws + off;
        off += bytes;
        return p;
    };
    _Float16* msgs   = (_Float16*)carve((size_t)n_edges * D_IN * sizeof(_Float16));
    float*    scores = (float*)   carve((size_t)n_edges * sizeof(float));
    float*    exbuf  = (float*)   carve((size_t)n_edges * sizeof(float));
    uint32_t* segmax = (uint32_t*)carve((size_t)n_nodes * sizeof(uint32_t));
    float*    denom  = (float*)   carve((size_t)n_nodes * sizeof(float));
    uint32_t* wfrag  = (uint32_t*)carve(6144 * sizeof(uint32_t));
    (void)ws_size; (void)n_in; (void)out_size;

    float* out      = (float*)d_out;
    float* attn_out = out + (size_t)n_nodes * D_IN;

    prep_frags<<<16, 256, 0, stream>>>(W1, W2, wfrag);

    int init_elems = n_nodes * D_IN;
    init_buffers<<<(init_elems + 255) / 256, 256, 0, stream>>>(out, segmax, denom, n_nodes);

    int tiles = (n_edges + 15) / 16;
    edge_mlp<<<(tiles + 7) / 8, 256, 0, stream>>>(x, src, edge_attr, wfrag,
                                                  b1, b2, attn_v, msgs, scores, n_edges);

    seg_max_k<<<(n_edges + 255) / 256, 256, 0, stream>>>(scores, tgt, segmax, n_edges);
    denom_k<<<(n_edges + 255) / 256, 256, 0, stream>>>(scores, tgt, segmax, exbuf, denom, n_edges);

    long total = (long)n_edges * D_IN;
    scatter_k<<<(int)((total + 255) / 256), 256, 0, stream>>>(msgs, tgt, exbuf, denom,
                                                              out, attn_out, total);
}
